// FiLM_49323404427978
// MI455X (gfx1250) — matrix operand order repaired and checked by reading, not compiled, not run on hardware
//
#include <hip/hip_runtime.h>
#include <hip/hip_bf16.h>
#include <hip/amd_detail/amd_hip_unsafe_atomics.h>

// ---------------- types ----------------
typedef __attribute__((ext_vector_type(16))) __bf16 bf16x16;
typedef __attribute__((ext_vector_type(8)))  float  v8f;

#define NN    50000
#define EE    1600000
#define CI    64

// ---------------- small utility kernels ----------------

__global__ void film_zero_f32(float* __restrict__ p, int n) {
    int i = blockIdx.x * blockDim.x + threadIdx.x;
    if (i < n) p[i] = 0.0f;
}

// deg[dst] += 1 per edge
__global__ void film_deg(const int* __restrict__ dst, float* __restrict__ deg, int e_cnt) {
    int e = blockIdx.x * blockDim.x + threadIdx.x;
    if (e < e_cnt) unsafeAtomicAdd(&deg[dst[e]], 1.0f);
}

// deg -> 1/max(deg,1)  (in place)
__global__ void film_deginv(float* __restrict__ deg, int n) {
    int i = blockIdx.x * blockDim.x + threadIdx.x;
    if (i < n) deg[i] = 1.0f / fmaxf(deg[i], 1.0f);
}

// fp32 x -> bf16 A-operand buffer (stride CI)
__global__ void film_cvt_x(const float* __restrict__ x, __bf16* __restrict__ xb, int total) {
    int i = blockIdx.x * blockDim.x + threadIdx.x;
    if (i < total) xb[i] = (__bf16)x[i];
}

// pack [W ; Fm ; Ws ; Fs] row-concatenated -> bf16 (6co x 64)
__global__ void film_pack_w(__bf16* __restrict__ wb,
                            const float* __restrict__ W,  const float* __restrict__ Fm,
                            const float* __restrict__ Ws, const float* __restrict__ Fs,
                            int co) {
    int idx = blockIdx.x * blockDim.x + threadIdx.x;
    int total = 6 * co * CI;
    if (idx >= total) return;
    int row = idx >> 6;      // /64
    int col = idx & 63;
    float v;
    if      (row <     co) v = W [(size_t)row            * CI + col];
    else if (row < 3 * co) v = Fm[(size_t)(row -     co) * CI + col];
    else if (row < 4 * co) v = Ws[(size_t)(row - 3 * co) * CI + col];
    else                   v = Fs[(size_t)(row - 4 * co) * CI + col];
    wb[idx] = (__bf16)v;
}

// ---------------- fused 4-GEMM via WMMA (bf16 -> f32) ----------------
// y[node][0:co]      = h      = x @ W.T
// y[node][co:2co]    = beta   (Fm rows 0..co-1)
// y[node][2co:3co]   = gamma  (Fm rows co..2co-1)
// y[node][3co:4co]   = hs     = x @ Ws.T
// y[node][4co:5co]   = beta_s
// y[node][5co:6co]   = gamma_s
__global__ __launch_bounds__(128)
void film_gemm(const __bf16* __restrict__ xb, const __bf16* __restrict__ wb,
               float* __restrict__ y, int rowsB) {
    const int lane    = threadIdx.x & 31;
    const int wave    = threadIdx.x >> 5;
    const int ntiles  = rowsB >> 4;
    const int nt      = blockIdx.y * 4 + wave;     // output-column tile
    if (nt >= ntiles) return;                      // wave-uniform branch
    const int mt      = blockIdx.x;                // node tile (16 nodes)
    const int m0      = mt << 4;
    const int mr      = lane & 15;
    const int hsel    = lane >> 4;                 // 0 or 1

    // A: row m0+mr, K = 8*hsel + {0..7} and 16 + 8*hsel + {0..7}, per K-step
    const __bf16* arow = xb + (size_t)(m0 + mr) * CI + 8 * hsel;
    // B: column nt*16+mr  == row of packed weights, K = 8*hsel + {0..7} and 16 + 8*hsel + {0..7}
    const __bf16* brow = wb + (size_t)(nt * 16 + mr) * CI + 8 * hsel;

    v8f acc = {};
#pragma unroll
    for (int kk = 0; kk < CI; kk += 32) {
        bf16x16 a, b;
#pragma unroll
        for (int j = 0; j < 8; ++j) {
            a[j]     = arow[kk + j];
            a[j + 8] = arow[kk + 16 + j];
        }
#pragma unroll
        for (int j = 0; j < 8; ++j) {
            b[j]     = brow[kk + j];
            b[j + 8] = brow[kk + 16 + j];
        }
        acc = __builtin_amdgcn_wmma_f32_16x16x32_bf16(
                  false, a, false, b, (short)0, acc, false, false);
    }

    const int n = nt * 16 + mr;
#pragma unroll
    for (int r = 0; r < 8; ++r) {
        int m = m0 + r + 8 * hsel;
        y[(size_t)m * rowsB + n] = acc[r];
    }
}

// ---------------- self path: out_self = relu(gamma_s*hs + beta_s); agg = 0 ----
__global__ void film_self(const float* __restrict__ y, float* __restrict__ out_self,
                          float* __restrict__ agg, int co) {
    const int n = blockIdx.x;
    const int c = threadIdx.x;         // blockDim.x == co
    const float* row = y + (size_t)n * 6 * co;
    float hs = row[3 * co + c];
    float bs = row[4 * co + c];
    float gs = row[5 * co + c];
    float v  = fmaf(gs, hs, bs);
    size_t idx = (size_t)n * co + c;
    out_self[idx] = v > 0.0f ? v : 0.0f;
    agg[idx]      = 0.0f;
}

// ---------------- edge message + segment-sum (atomics) ----------------
__global__ __launch_bounds__(64)
void film_edge(const float* __restrict__ y, const int* __restrict__ src,
               const int* __restrict__ dst, float* __restrict__ agg, int co) {
    const int e = blockIdx.x;
    const int c = threadIdx.x;         // blockDim.x == 64 (>= co)
    if (c >= co) return;
    const int s = src[e];
    const int d = dst[e];
    const float* rs = y + (size_t)s * 6 * co;   // h at [0,co)
    const float* rd = y + (size_t)d * 6 * co;
    float msg = fmaf(rd[2 * co + c], rs[c], rd[co + c]);  // gamma*h + beta
    if (msg > 0.0f)                                        // relu: skip zero adds
        unsafeAtomicAdd(&agg[(size_t)d * co + c], msg);
}

// ---------------- combine: h_next = relu(out_self + deg_inv*agg) --------------
__global__ void film_combine(const float* __restrict__ out_self,
                             const float* __restrict__ agg,
                             const float* __restrict__ deg_inv,
                             __bf16* __restrict__ xb_next,
                             float* __restrict__ hfin,
                             int co, int write_f32) {
    const int n = blockIdx.x;
    const int c = threadIdx.x;         // blockDim.x == co
    size_t idx = (size_t)n * co + c;
    float v = fmaf(deg_inv[n], agg[idx], out_self[idx]);
    v = v > 0.0f ? v : 0.0f;
    if (write_f32) hfin[idx] = v;
    else           xb_next[(size_t)n * CI + c] = (__bf16)v;
}

// ---------------- log_softmax over 40 channels ----------------
__global__ void film_logsoftmax(const float* __restrict__ h, float* __restrict__ out,
                                int n_nodes, int co) {
    int n = blockIdx.x * blockDim.x + threadIdx.x;
    if (n >= n_nodes) return;
    const float* r = h + (size_t)n * co;
    float mx = -3.4e38f;
    for (int c = 0; c < co; ++c) mx = fmaxf(mx, r[c]);
    float s = 0.0f;
    for (int c = 0; c < co; ++c) s += __expf(r[c] - mx);
    float l = __logf(s);
    float* o = out + (size_t)n * co;
    for (int c = 0; c < co; ++c) o[c] = r[c] - mx - l;
}

// ---------------- host side ----------------
extern "C" void kernel_launch(void* const* d_in, const int* in_sizes, int n_in,
                              void* d_out, int out_size, void* d_ws, size_t ws_size,
                              hipStream_t stream) {
    (void)in_sizes; (void)n_in; (void)out_size; (void)ws_size;

    const float* x   = (const float*)d_in[0];
    const int*   ei  = (const int*)d_in[1];
    const int*   src = ei;
    const int*   dst = ei + EE;
    const float* W [3] = { (const float*)d_in[3],  (const float*)d_in[7],  (const float*)d_in[11] };
    const float* Fm[3] = { (const float*)d_in[4],  (const float*)d_in[8],  (const float*)d_in[12] };
    const float* Ws[3] = { (const float*)d_in[5],  (const float*)d_in[9],  (const float*)d_in[13] };
    const float* Fs[3] = { (const float*)d_in[6],  (const float*)d_in[10], (const float*)d_in[14] };

    // workspace carve (256B aligned)
    char* ws = (char*)d_ws;
    size_t off = 0;
    auto carve = [&](size_t bytes) {
        char* p = ws + off;
        off += (bytes + 255) & ~(size_t)255;
        return p;
    };
    __bf16* xb       = (__bf16*)carve((size_t)NN * CI * sizeof(__bf16));
    __bf16* wb       = (__bf16*)carve((size_t)6 * CI * CI * sizeof(__bf16));
    float*  yall     = (float*) carve((size_t)NN * 6 * CI * sizeof(float));
    float*  out_self = (float*) carve((size_t)NN * CI * sizeof(float));
    float*  agg      = (float*) carve((size_t)NN * CI * sizeof(float));
    float*  deg      = (float*) carve((size_t)NN * sizeof(float));
    float*  hfin     = (float*) carve((size_t)NN * 40 * sizeof(float));

    // degree -> 1/max(deg,1), in place
    film_zero_f32<<<(NN + 255) / 256, 256, 0, stream>>>(deg, NN);
    film_deg<<<(EE + 255) / 256, 256, 0, stream>>>(dst, deg, EE);
    film_deginv<<<(NN + 255) / 256, 256, 0, stream>>>(deg, NN);

    // x -> bf16
    film_cvt_x<<<(NN * CI + 255) / 256, 256, 0, stream>>>(x, xb, NN * CI);

    for (int l = 0; l < 3; ++l) {
        const int co    = (l == 2) ? 40 : 64;
        const int rowsB = 6 * co;

        film_pack_w<<<(rowsB * CI + 255) / 256, 256, 0, stream>>>(
            wb, W[l], Fm[l], Ws[l], Fs[l], co);

        dim3 ggrid(NN / 16, (rowsB / 16 + 3) / 4);
        film_gemm<<<ggrid, 128, 0, stream>>>(xb, wb, yall, rowsB);

        film_self<<<NN, co, 0, stream>>>(yall, out_self, agg, co);

        film_edge<<<EE, 64, 0, stream>>>(yall, src, dst, agg, co);

        film_combine<<<NN, co, 0, stream>>>(out_self, agg, deg, xb, hfin,
                                            co, (l == 2) ? 1 : 0);
    }

    film_logsoftmax<<<(NN + 255) / 256, 256, 0, stream>>>(hfin, (float*)d_out, NN, 40);
}
